// DistributedPiKVMoE_48069273977258
// MI455X (gfx1250) — matrix operand order repaired and checked by reading, not compiled, not run on hardware
//
#include <hip/hip_runtime.h>
#include <math.h>

// ---------------------------------------------------------------------------
// Types for CDNA5 WMMA (wave32): v_wmma_f32_16x16x32_bf16
// ---------------------------------------------------------------------------
typedef __attribute__((ext_vector_type(16))) __bf16 v16bf;
typedef __attribute__((ext_vector_type(8)))  float  v8f;

struct __align__(16) Chunk { unsigned int u[4]; };   // 16-byte (8 x bf16) chunk
union Frag { v16bf v; Chunk c[2]; };                 // 32 bytes = 8 VGPRs

__device__ __forceinline__ unsigned short f32_to_bf16(float f) {
    unsigned int u = __float_as_uint(f);
    u = (u + 0x7FFFu + ((u >> 16) & 1u)) >> 16;      // round-to-nearest-even
    return (unsigned short)u;
}

// CDNA5 async global->LDS copy (16B per lane), tracked by ASYNCcnt.
__device__ __forceinline__ void async_copy_b128(const unsigned short* gsrc,
                                                const unsigned short* lds_dst) {
    // Low 32 bits of a flat LDS pointer are the LDS byte offset (ISA 10.2).
    unsigned loff = (unsigned)(unsigned long long)lds_dst;
    asm volatile("global_load_async_to_lds_b128 %0, %1, off"
                 :: "v"(loff), "v"(gsrc) : "memory");
}
__device__ __forceinline__ void async_wait4() {
    asm volatile("s_wait_asynccnt 0x4" ::: "memory");
}
__device__ __forceinline__ void async_wait0() {
    asm volatile("s_wait_asynccnt 0x0" ::: "memory");
}

// ---------------------------------------------------------------------------
// Problem constants
// ---------------------------------------------------------------------------
#define BB   4
#define SS   1024
#define HH   512
#define EE   8
#define FF   2048
#define TOK  (BB * SS)          // 4096 tokens

// GEMM tiling
#define BM 128
#define BN 128
#define BK 32
#define LDSK 40                 // padded LDS row stride (bf16 elems)
#define AELEMS (BM * LDSK)      // A tile size in elements
#define BUFSZ  ((BM + BN) * LDSK)

// ---------------------------------------------------------------------------
// Tiled bf16 WMMA GEMM: C = A[M,K] * B^T  where Bm is stored [N][K] (ldb = K)
// Double-buffered LDS, async global->LDS staging (ASYNCcnt pipeline),
// branch-free steady-state K-loop (last iteration peeled).
// MODE epilogues:
//  0: Cf[m,n] = acc*scale                    (QK^T scores)
//  1: Cb[m,n] = bf16(acc)                    (attn @ V)
//  2: Cb[m,n] = bf16(acc + bias[n])          (Q,K projections)
//  3: Cb[m,n] = bf16(gelu(acc + bias[n]))    (fc1, exact erf GELU)
//  4: Cf[m,n] = acc + bias[n]                (fc2 -> FFN fp32)
//  5: Cb[n,m] = bf16(acc + bias[n])          (V projection, stored transposed)
//  6: Cf[m,n] += gate[m*8]*(acc+bias[n]+extra[m,n])   (O proj, final combine)
// ---------------------------------------------------------------------------
template<int MODE>
__global__ __launch_bounds__(256)
void gemm_wmma_bf16(const unsigned short* __restrict__ A,
                    const unsigned short* __restrict__ Bm,
                    long sA, long sB, long sC,
                    float* __restrict__ Cf, unsigned short* __restrict__ Cb,
                    int M, int N, int K, int lda, int ldc,
                    const float* __restrict__ bias,
                    const float* __restrict__ extra,
                    const float* __restrict__ gate,
                    float scale)
{
    // [buf0: A(128x40) | B(128x40)][buf1: A | B]   = 40 KB
    __shared__ unsigned short smem[2 * BUFSZ];

    const int tid  = threadIdx.x;
    const int lane = tid & 31;
    const int wid  = tid >> 5;
    const int wm   = wid >> 2;            // 0..1 : 64-row slab
    const int wn   = wid & 3;             // 0..3 : 32-col slab
    const int z    = blockIdx.z;
    const int m0   = blockIdx.y * BM;
    const int n0   = blockIdx.x * BN;

    A  += (long)z * sA;
    Bm += (long)z * sB;

    v8f acc[4][2];
    #pragma unroll
    for (int i = 0; i < 4; ++i)
        #pragma unroll
        for (int j = 0; j < 2; ++j)
            #pragma unroll
            for (int l = 0; l < 8; ++l) acc[i][j][l] = 0.0f;

    // Each wave stages 4 async b128 ops per tile (2 A-chunks + 2 B-chunks/thread).
    auto issue_tile = [&](int kk, int buf) {
        const int kbase = kk * BK;
        #pragma unroll
        for (int i = 0; i < 2; ++i) {
            const int ci  = tid + i * 256;        // 0..511
            const int row = ci >> 2;              // 0..127
            const int cc  = (ci & 3) * 8;         // 0,8,16,24
            async_copy_b128(A  + (long)(m0 + row) * lda + kbase + cc,
                            &smem[buf * BUFSZ + row * LDSK + cc]);
            async_copy_b128(Bm + (long)(n0 + row) * K + kbase + cc,
                            &smem[buf * BUFSZ + AELEMS + row * LDSK + cc]);
        }
    };

    const int ml  = lane & 15;
    const int akb = (lane < 16) ? 0 : 8;    // A: K pair base (ISA layout)
    const int bkb = (lane < 16) ? 0 : 8;    // B: K pair base (ISA layout)

    auto do_tile = [&](int buf) {
        const unsigned short* Asb = &smem[buf * BUFSZ];
        const unsigned short* Bsb = &smem[buf * BUFSZ + AELEMS];
        Frag bfr[2];
        #pragma unroll
        for (int ni = 0; ni < 2; ++ni) {
            const int n = wn * 32 + ni * 16 + ml;
            bfr[ni].c[0] = *(const Chunk*)(&Bsb[n * LDSK + bkb]);
            bfr[ni].c[1] = *(const Chunk*)(&Bsb[n * LDSK + bkb + 16]);
        }
        #pragma unroll
        for (int mi = 0; mi < 4; ++mi) {
            const int m = wm * 64 + mi * 16 + ml;
            Frag afr;
            afr.c[0] = *(const Chunk*)(&Asb[m * LDSK + akb]);
            afr.c[1] = *(const Chunk*)(&Asb[m * LDSK + akb + 16]);
            #pragma unroll
            for (int ni = 0; ni < 2; ++ni) {
                acc[mi][ni] = __builtin_amdgcn_wmma_f32_16x16x32_bf16(
                    false, afr.v, false, bfr[ni].v,
                    (short)0, acc[mi][ni], false, false);
            }
        }
    };

    const int nk = K / BK;     // >= 16 for all GEMMs here
    issue_tile(0, 0);

    // Steady state: branch-free (last iteration peeled).
    for (int kk = 0; kk < nk - 1; ++kk) {
        issue_tile(kk + 1, (kk + 1) & 1);  // overlap next fetch with this WMMA
        async_wait4();                     // tile kk's 4 ops done (in-order)
        __syncthreads();
        do_tile(kk & 1);
        __syncthreads();                   // protect buf before kk+2 overwrites
    }
    // Final tile.
    async_wait0();
    __syncthreads();
    do_tile((nk - 1) & 1);

    // ---- epilogue ----
    const int msel = (lane < 16) ? 0 : 8;
    #pragma unroll
    for (int mi = 0; mi < 4; ++mi) {
        #pragma unroll
        for (int ni = 0; ni < 2; ++ni) {
            const int ng = n0 + wn * 32 + ni * 16 + ml;
            #pragma unroll
            for (int v = 0; v < 8; ++v) {
                const int mg = m0 + wm * 64 + mi * 16 + msel + v;
                float val = acc[mi][ni][v];
                if (MODE == 0) {
                    Cf[(long)z * sC + (long)mg * ldc + ng] = val * scale;
                } else if (MODE == 1) {
                    Cb[(long)z * sC + (long)mg * ldc + ng] = f32_to_bf16(val);
                } else if (MODE == 2) {
                    Cb[(long)z * sC + (long)mg * ldc + ng] =
                        f32_to_bf16(val + bias[ng]);
                } else if (MODE == 3) {
                    float t = val + bias[ng];
                    t = 0.5f * t * (1.0f + erff(t * 0.70710678118654752f));
                    Cb[(long)z * sC + (long)mg * ldc + ng] = f32_to_bf16(t);
                } else if (MODE == 4) {
                    Cf[(long)z * sC + (long)mg * ldc + ng] = val + bias[ng];
                } else if (MODE == 5) {
                    Cb[(long)z * sC + (long)ng * ldc + mg] =
                        f32_to_bf16(val + bias[ng]);
                } else { // MODE == 6
                    const float t = val + bias[ng] + extra[(long)mg * ldc + ng];
                    Cf[(long)mg * ldc + ng] += gate[(long)mg * 8] * t;
                }
            }
        }
    }
}

// ---------------------------------------------------------------------------
// Small helper kernels
// ---------------------------------------------------------------------------
__global__ void zero_f32(float* p, long n) {
    long i = (long)blockIdx.x * 256 + threadIdx.x;
    if (i < n) p[i] = 0.0f;
}

__global__ void cast_bf16(const float* __restrict__ s, unsigned short* __restrict__ d, long n) {
    long i = (long)blockIdx.x * 256 + threadIdx.x;
    if (i < n) d[i] = f32_to_bf16(s[i]);
}

// src: [z][K][N] fp32  ->  dst: [z][N][K] bf16
__global__ void transpose_cast(const float* __restrict__ src,
                               unsigned short* __restrict__ dst, int K, int N) {
    const long kn   = (long)K * N;
    const long base = (long)blockIdx.z * kn;
    const long idx  = (long)blockIdx.x * 256 + threadIdx.x;
    if (idx >= kn) return;
    const long k = idx / N, n = idx % N;
    dst[base + n * (long)K + k] = f32_to_bf16(src[base + idx]);
}

// logits[t][e] = x[t,:].Wg[:,e] + bg[e]   (tiny, VALU)
__global__ void router_logits(const float* __restrict__ x, const float* __restrict__ Wg,
                              const float* __restrict__ bg, float* __restrict__ logits) {
    const int t = blockIdx.x * 32 + (threadIdx.x >> 3);
    const int e = threadIdx.x & 7;
    const float* xr = x + (long)t * HH;
    float s = 0.0f;
    for (int h = 0; h < HH; ++h) s += xr[h] * Wg[h * EE + e];
    logits[t * EE + e] = s + bg[e];
}

// per-token softmax over 8 experts, top-2 renormalized gate, lb-loss partials
__global__ void router_gate(const float* __restrict__ logits,
                            float* __restrict__ gate, float* __restrict__ red) {
    const int t = blockIdx.x * 256 + threadIdx.x;
    if (t >= TOK) return;
    float p[EE];
    float mx = -1e30f;
    for (int e = 0; e < EE; ++e) { p[e] = logits[t * EE + e]; mx = fmaxf(mx, p[e]); }
    float sum = 0.0f;
    for (int e = 0; e < EE; ++e) { p[e] = expf(p[e] - mx); sum += p[e]; }
    const float inv = 1.0f / sum;
    for (int e = 0; e < EE; ++e) { p[e] *= inv; atomicAdd(&red[EE + e], p[e]); }
    int i0 = 0;
    for (int e = 1; e < EE; ++e) if (p[e] > p[i0]) i0 = e;
    int i1 = -1;
    for (int e = 0; e < EE; ++e) {
        if (e == i0) continue;
        if (i1 < 0 || p[e] > p[i1]) i1 = e;
    }
    const float ws = p[i0] + p[i1];
    for (int e = 0; e < EE; ++e) gate[t * EE + e] = 0.0f;
    gate[t * EE + i0] = p[i0] / ws;
    gate[t * EE + i1] = p[i1] / ws;
    atomicAdd(&red[i0], 1.0f);
    atomicAdd(&red[i1], 1.0f);
}

__global__ void lb_loss(const float* __restrict__ red, float* __restrict__ out_loss) {
    if (threadIdx.x == 0 && blockIdx.x == 0) {
        float s = 0.0f;
        for (int e = 0; e < EE; ++e)
            s += (red[e] / (float)TOK) * (red[EE + e] / (float)TOK);
        out_loss[0] = (float)EE * s;
    }
}

// row softmax: scores fp32 [4][1024][1024] -> attn bf16
__global__ __launch_bounds__(256)
void softmax_row(const float* __restrict__ sc, unsigned short* __restrict__ attn) {
    __shared__ float red[256];
    const long row = (long)blockIdx.z * SS + blockIdx.x;
    const float* r = sc + row * SS;
    unsigned short* o = attn + row * SS;
    const int t = threadIdx.x;
    float mx = -1e30f;
    for (int i = t; i < SS; i += 256) mx = fmaxf(mx, r[i]);
    red[t] = mx; __syncthreads();
    for (int s = 128; s > 0; s >>= 1) { if (t < s) red[t] = fmaxf(red[t], red[t + s]); __syncthreads(); }
    mx = red[0]; __syncthreads();
    float sum = 0.0f;
    for (int i = t; i < SS; i += 256) sum += expf(r[i] - mx);
    red[t] = sum; __syncthreads();
    for (int s = 128; s > 0; s >>= 1) { if (t < s) red[t] += red[t + s]; __syncthreads(); }
    const float inv = 1.0f / red[0];
    for (int i = t; i < SS; i += 256) o[i] = f32_to_bf16(expf(r[i] - mx) * inv);
}

// ---------------------------------------------------------------------------
// Host launcher
// ---------------------------------------------------------------------------
extern "C" void kernel_launch(void* const* d_in, const int* in_sizes, int n_in,
                              void* d_out, int out_size, void* d_ws, size_t ws_size,
                              hipStream_t stream) {
    (void)in_sizes; (void)n_in; (void)out_size; (void)ws_size;

    const float* x     = (const float*)d_in[0];
    const float* Wg    = (const float*)d_in[1];
    const float* bg    = (const float*)d_in[2];
    const float* fc1_w = (const float*)d_in[3];
    const float* fc1_b = (const float*)d_in[4];
    const float* fc2_w = (const float*)d_in[5];
    const float* fc2_b = (const float*)d_in[6];
    const float* q_w   = (const float*)d_in[7];
    const float* q_b   = (const float*)d_in[8];
    const float* k_w   = (const float*)d_in[9];
    const float* k_b   = (const float*)d_in[10];
    const float* v_w   = (const float*)d_in[11];
    const float* v_b   = (const float*)d_in[12];
    const float* o_w   = (const float*)d_in[13];
    const float* o_b   = (const float*)d_in[14];

    float* out = (float*)d_out;                       // [4096][512] + loss at end

    // -------- workspace layout (bytes) --------
    char*  ws  = (char*)d_ws;
    size_t off = 0;
    auto alloc = [&](size_t bytes) { char* p = ws + off; off += (bytes + 255) & ~(size_t)255; return p; };

    unsigned short* WT1 = (unsigned short*)alloc((size_t)EE * HH * FF * 2);   // [E][F][H]
    unsigned short* WT2 = (unsigned short*)alloc((size_t)EE * FF * HH * 2);   // [E][H][F]
    unsigned short* WTQ = (unsigned short*)alloc((size_t)EE * HH * HH * 2);
    unsigned short* WTK = (unsigned short*)alloc((size_t)EE * HH * HH * 2);
    unsigned short* WTV = (unsigned short*)alloc((size_t)EE * HH * HH * 2);
    unsigned short* WTO = (unsigned short*)alloc((size_t)EE * HH * HH * 2);
    unsigned short* XB  = (unsigned short*)alloc((size_t)TOK * HH * 2);
    unsigned short* H1  = (unsigned short*)alloc((size_t)TOK * FF * 2);
    float*          FFN = (float*)         alloc((size_t)TOK * HH * 4);
    unsigned short* QB  = (unsigned short*)alloc((size_t)TOK * HH * 2);
    unsigned short* KB  = (unsigned short*)alloc((size_t)TOK * HH * 2);
    unsigned short* VT  = (unsigned short*)alloc((size_t)BB * HH * SS * 2);   // [b][H][S]
    unsigned short* AB  = (unsigned short*)alloc((size_t)TOK * HH * 2);
    float*          SC  = (float*)         alloc((size_t)BB * SS * SS * 4);
    unsigned short* AT  = (unsigned short*)alloc((size_t)BB * SS * SS * 2);
    float*          LG  = (float*)         alloc((size_t)TOK * EE * 4);
    float*          GT  = (float*)         alloc((size_t)TOK * EE * 4);
    float*          RED = (float*)         alloc(64);

    // -------- setup: zero output + reduction scratch, cast/transpose --------
    {
        const long nout = (long)TOK * HH;
        zero_f32<<<dim3((nout + 255) / 256), 256, 0, stream>>>(out, nout);
        zero_f32<<<dim3(1), 256, 0, stream>>>(RED, 16);

        const long nx = (long)TOK * HH;
        cast_bf16<<<dim3((nx + 255) / 256), 256, 0, stream>>>(x, XB, nx);

        transpose_cast<<<dim3((HH * FF + 255) / 256, 1, EE), 256, 0, stream>>>(fc1_w, WT1, HH, FF);
        transpose_cast<<<dim3((FF * HH + 255) / 256, 1, EE), 256, 0, stream>>>(fc2_w, WT2, FF, HH);
        transpose_cast<<<dim3((HH * HH + 255) / 256, 1, EE), 256, 0, stream>>>(q_w, WTQ, HH, HH);
        transpose_cast<<<dim3((HH * HH + 255) / 256, 1, EE), 256, 0, stream>>>(k_w, WTK, HH, HH);
        transpose_cast<<<dim3((HH * HH + 255) / 256, 1, EE), 256, 0, stream>>>(v_w, WTV, HH, HH);
        transpose_cast<<<dim3((HH * HH + 255) / 256, 1, EE), 256, 0, stream>>>(o_w, WTO, HH, HH);
    }

    // -------- router --------
    router_logits<<<dim3(TOK / 32), 256, 0, stream>>>(x, Wg, bg, LG);
    router_gate<<<dim3(TOK / 256), 256, 0, stream>>>(LG, GT, RED);

    const float scale = 0.044194173824159216f;   // 1/sqrt(512)

    // -------- per-expert dense forward, accumulated into out --------
    for (int e = 0; e < EE; ++e) {
        const unsigned short* w1 = WT1 + (size_t)e * FF * HH;   // [F][H]
        const unsigned short* w2 = WT2 + (size_t)e * HH * FF;   // [H][F]
        const unsigned short* wq = WTQ + (size_t)e * HH * HH;
        const unsigned short* wk = WTK + (size_t)e * HH * HH;
        const unsigned short* wv = WTV + (size_t)e * HH * HH;
        const unsigned short* wo = WTO + (size_t)e * HH * HH;

        // fc1 + exact GELU -> H1 bf16      [4096,2048] = [4096,512]x[512,2048]
        gemm_wmma_bf16<3><<<dim3(FF / BN, TOK / BM, 1), 256, 0, stream>>>(
            XB, w1, 0, 0, 0, nullptr, H1, TOK, FF, HH, HH, FF,
            fc1_b + (size_t)e * FF, nullptr, nullptr, 1.0f);

        // fc2 -> FFN fp32                  [4096,512] = [4096,2048]x[2048,512]
        gemm_wmma_bf16<4><<<dim3(HH / BN, TOK / BM, 1), 256, 0, stream>>>(
            H1, w2, 0, 0, 0, FFN, nullptr, TOK, HH, FF, FF, HH,
            fc2_b + (size_t)e * HH, nullptr, nullptr, 1.0f);

        // Q, K projections -> bf16
        gemm_wmma_bf16<2><<<dim3(HH / BN, TOK / BM, 1), 256, 0, stream>>>(
            XB, wq, 0, 0, 0, nullptr, QB, TOK, HH, HH, HH, HH,
            q_b + (size_t)e * HH, nullptr, nullptr, 1.0f);
        gemm_wmma_bf16<2><<<dim3(HH / BN, TOK / BM, 1), 256, 0, stream>>>(
            XB, wk, 0, 0, 0, nullptr, KB, TOK, HH, HH, HH, HH,
            k_b + (size_t)e * HH, nullptr, nullptr, 1.0f);

        // V projection stored transposed per batch: VT[b][h][s]
        gemm_wmma_bf16<5><<<dim3(HH / BN, SS / BM, BB), 256, 0, stream>>>(
            XB, wv, (long)SS * HH, 0, (long)HH * SS, nullptr, VT,
            SS, HH, HH, HH, SS, v_b + (size_t)e * HH, nullptr, nullptr, 1.0f);

        // scores = Q K^T * scale (batched): [1024,1024] = [1024,512]x[512,1024]
        gemm_wmma_bf16<0><<<dim3(SS / BN, SS / BM, BB), 256, 0, stream>>>(
            QB, KB, (long)SS * HH, (long)SS * HH, (long)SS * SS, SC, nullptr,
            SS, SS, HH, HH, SS, nullptr, nullptr, nullptr, scale);

        // softmax rows -> bf16 attn
        softmax_row<<<dim3(SS, 1, BB), 256, 0, stream>>>(SC, AT);

        // a = attn @ V (batched): [1024,512] = [1024,1024]x[1024,512]
        gemm_wmma_bf16<1><<<dim3(HH / BN, SS / BM, BB), 256, 0, stream>>>(
            AT, VT, (long)SS * SS, (long)HH * SS, (long)SS * HH, nullptr, AB,
            SS, HH, SS, SS, HH, nullptr, nullptr, nullptr, 1.0f);

        // out += gate_e * (a @ o_w + o_b + ffn)
        gemm_wmma_bf16<6><<<dim3(HH / BN, TOK / BM, 1), 256, 0, stream>>>(
            AB, wo, 0, 0, 0, out, nullptr, TOK, HH, HH, HH, HH,
            o_b + (size_t)e * HH, FFN, GT + e, 1.0f);
    }

    // -------- load-balance loss -> last output element --------
    lb_loss<<<dim3(1), 64, 0, stream>>>(RED, out + (size_t)TOK * HH);
}
